// MoE_80573586473652
// MI455X (gfx1250) — matrix operand order repaired and checked by reading, not compiled, not run on hardware
//
#include <hip/hip_runtime.h>
#include <math.h>

// ---------------- problem constants (match reference) ----------------
#define B_TOK 4096
#define D_IN  1024
#define O_OUT 1024
#define H_HID 4096
#define N_EXP 8
#define LOSS_COEF 0.01f

// ---------------- WMMA vector types (gfx1250, wave32) ----------------
typedef __attribute__((ext_vector_type(4)))  __bf16 v4bf;
typedef __attribute__((ext_vector_type(8)))  float  v8f;
typedef __attribute__((ext_vector_type(8)))  __bf16 v8bf;
typedef __attribute__((ext_vector_type(16))) __bf16 v16bf;
typedef __attribute__((ext_vector_type(4)))  float  v4f;

// ---------------- tiling ----------------
#define TT    32            // tokens per block tile (2 WMMA m-tiles)
#define HC    256           // H chunk fused through LDS (16 waves x 16 cols)
#define XS    (D_IN + 8)    // LDS x row stride (bf16 elems), padded
#define HS    (HC + 8)      // LDS h row stride (bf16 elems), padded
#define NWAVE 16
#define BLK   512
#define NCOLW 64            // output columns owned per wave (4 n-frags)

// ---------------- workspace layout (bytes) ----------------
#define WS_COUNTS 0                               // int[8]
#define WS_IMP    64                              // float[8]
#define WS_LOAD   128                             // float[8]
#define WS_TOK    256                             // int[N_EXP*B_TOK]
#define WS_GATE   (256 + N_EXP * B_TOK * 4)       // float[N_EXP*B_TOK]
#define WS_W1T    262400                          // bf16 [E][H][D] (64 MB)
#define WS_W2T    (WS_W1T + (size_t)N_EXP * H_HID * D_IN * 2)
#define WS_NEEDED (WS_W2T + (size_t)N_EXP * O_OUT * H_HID * 2)

// dynamic LDS layout (bytes) for expert kernel
#define SM_X    0
#define SM_H    (SM_X + TT * XS * 2)              // 66048
#define SM_RED  (SM_H + TT * HS * 2)              // + 16896 = 82944
#define SM_ROW  (SM_RED + TT * NWAVE * 4)         // + 2048  = 84992
#define SM_TOK  (SM_ROW + TT * 4)                 // 85120
#define SM_GATE (SM_TOK + TT * 4)                 // 85248
#define SM_SIZE (SM_GATE + TT * 4)                // 85376

// ---------------------------------------------------------------------
// helpers
// ---------------------------------------------------------------------
__device__ __forceinline__ v8f wmma_bf16(v16bf a, v16bf b, v8f c) {
    return __builtin_amdgcn_wmma_f32_16x16x32_bf16(
        /*neg_a=*/false, a, /*neg_b=*/false, b,
        /*c_mod=*/(short)0, c, /*reuse_a=*/false, /*reuse_b=*/false);
}

// A fragment (16x32 bf16, MxK) from LDS. Lane L: M = L&15,
// K = base..base+15 (elems 0..15),
// base = k0 + 16*(L>>4).  Two 16-byte ds loads.
__device__ __forceinline__ v16bf load_a_frag(const __bf16* lds, int stride,
                                             int mBase, int k0, int half, int c) {
    const __bf16* p = lds + (size_t)(mBase + c) * stride + k0 + half * 16;
    v8bf lo = *reinterpret_cast<const v8bf*>(p);
    v8bf hi = *reinterpret_cast<const v8bf*>(p + 8);
    v16bf a;
#pragma unroll
    for (int j = 0; j < 8; ++j) { a[j] = lo[j]; a[j + 8] = hi[j]; }
    return a;
}

// B fragment from f32 row-major weights (fallback path): 16 strided dword
// loads, converted to bf16.
__device__ __forceinline__ v16bf load_b_frag(const float* __restrict__ W,
                                             int ldw, int krow, int col) {
    const float* p = W + (size_t)krow * ldw + col;
    v16bf b;
#pragma unroll
    for (int j = 0; j < 16; ++j) b[j] = (__bf16)p[(size_t)j * ldw];
    return b;
}

// B fragment from pre-transposed bf16 weights WT[col][k] (ld = K extent):
// 16 contiguous K values per lane -> two global_load_b128, no converts.
__device__ __forceinline__ v16bf load_b_frag_T(const __bf16* __restrict__ WT,
                                               int ldk, int krow, int col) {
    const __bf16* p = WT + (size_t)col * ldk + krow;
    v8bf lo = *reinterpret_cast<const v8bf*>(p);
    v8bf hi = *reinterpret_cast<const v8bf*>(p + 8);
    v16bf b;
#pragma unroll
    for (int j = 0; j < 8; ++j) { b[j] = lo[j]; b[j + 8] = hi[j]; }
    return b;
}

// ---------------------------------------------------------------------
// kernel 0: zero y + routing counters
// ---------------------------------------------------------------------
__global__ void moe_init_kernel(float* __restrict__ y, int n,
                                int* counts, float* imp, float* loadv) {
    int idx = blockIdx.x * blockDim.x + threadIdx.x;
    for (int i = idx; i < n; i += gridDim.x * blockDim.x) y[i] = 0.0f;
    if (idx < N_EXP) { counts[idx] = 0; imp[idx] = 0.0f; loadv[idx] = 0.0f; }
}

// ---------------------------------------------------------------------
// kernel 0b: 32x32 LDS tile transpose + f32->bf16 convert
// src: [z][R][C] f32 row-major  ->  dst: [z][C][R] bf16 row-major
// block (32,8), each thread covers 4 tile rows
// ---------------------------------------------------------------------
__global__ void transpose_cvt_kernel(const float* __restrict__ src,
                                     __bf16* __restrict__ dst, int R, int C) {
    __shared__ float tile[32][33];
    const size_t eoff = (size_t)blockIdx.z * R * C;
    const int cbase = blockIdx.x * 32;
    const int rbase = blockIdx.y * 32;
    const int tx = threadIdx.x, ty = threadIdx.y;
#pragma unroll
    for (int i = 0; i < 32; i += 8)
        tile[ty + i][tx] = src[eoff + (size_t)(rbase + ty + i) * C + cbase + tx];
    __syncthreads();
#pragma unroll
    for (int i = 0; i < 32; i += 8)
        dst[eoff + (size_t)(cbase + ty + i) * R + rbase + tx] =
            (__bf16)tile[tx][ty + i];
}

// ---------------------------------------------------------------------
// kernel 1: gating — one wave32 per token
// ---------------------------------------------------------------------
__global__ void moe_gate_kernel(const float* __restrict__ x,
                                const float* __restrict__ wg,
                                int* counts, float* imp, float* loadv,
                                int* tok_list, float* gate_list) {
    const int wave = threadIdx.x >> 5;
    const int lane = threadIdx.x & 31;
    const int t = blockIdx.x * (blockDim.x >> 5) + wave;
    if (t >= B_TOK) return;

    float s[N_EXP];
#pragma unroll
    for (int e = 0; e < N_EXP; ++e) s[e] = 0.0f;

    const float* xr = x + (size_t)t * D_IN;
    for (int d = lane; d < D_IN; d += 32) {
        float xv = xr[d];
        const float* wr = wg + (size_t)d * N_EXP;
#pragma unroll
        for (int e = 0; e < N_EXP; ++e) s[e] += xv * wr[e];
    }
#pragma unroll
    for (int off = 16; off >= 1; off >>= 1) {
#pragma unroll
        for (int e = 0; e < N_EXP; ++e) s[e] += __shfl_xor(s[e], off, 32);
    }

    if (lane == 0) {
        float mx = s[0];
#pragma unroll
        for (int e = 1; e < N_EXP; ++e) mx = fmaxf(mx, s[e]);
        float p[N_EXP];
        float sum = 0.0f;
#pragma unroll
        for (int e = 0; e < N_EXP; ++e) { p[e] = __expf(s[e] - mx); sum += p[e]; }
        float inv = 1.0f / sum;
#pragma unroll
        for (int e = 0; e < N_EXP; ++e) p[e] *= inv;

        // top-2, strict > so ties pick the lower index (matches lax.top_k)
        int i1 = 0; float v1 = p[0];
#pragma unroll
        for (int e = 1; e < N_EXP; ++e) if (p[e] > v1) { v1 = p[e]; i1 = e; }
        int i2 = (i1 == 0) ? 1 : 0; float v2 = p[i2];
#pragma unroll
        for (int e = 0; e < N_EXP; ++e)
            if (e != i1 && p[e] > v2) { v2 = p[e]; i2 = e; }

        float denom = v1 + v2 + 1e-6f;
        float g1 = v1 / denom, g2 = v2 / denom;

        atomicAdd(&imp[i1], g1);
        atomicAdd(&imp[i2], g2);
        atomicAdd(&loadv[i1], 1.0f);
        atomicAdd(&loadv[i2], 1.0f);
        int s1 = atomicAdd(&counts[i1], 1);
        tok_list[i1 * B_TOK + s1] = t;  gate_list[i1 * B_TOK + s1] = g1;
        int s2 = atomicAdd(&counts[i2], 1);
        tok_list[i2 * B_TOK + s2] = t;  gate_list[i2 * B_TOK + s2] = g2;
    }
}

// ---------------------------------------------------------------------
// kernel 2: load-balance loss (scalar)
// ---------------------------------------------------------------------
__global__ void moe_loss_kernel(const float* __restrict__ imp,
                                const float* __restrict__ loadv,
                                float* __restrict__ out_loss) {
    if (threadIdx.x == 0 && blockIdx.x == 0) {
        float mi = 0.0f, ml = 0.0f;
#pragma unroll
        for (int e = 0; e < N_EXP; ++e) { mi += imp[e]; ml += loadv[e]; }
        mi *= (1.0f / N_EXP); ml *= (1.0f / N_EXP);
        float vi = 0.0f, vl = 0.0f;
#pragma unroll
        for (int e = 0; e < N_EXP; ++e) {
            float di = imp[e] - mi;   vi += di * di;
            float dl = loadv[e] - ml; vl += dl * dl;
        }
        vi *= (1.0f / (N_EXP - 1)); vl *= (1.0f / (N_EXP - 1));
        float cvi = vi / (mi * mi + 1e-10f);
        float cvl = vl / (ml * ml + 1e-10f);
        *out_loss = LOSS_COEF * (cvi + cvl);
    }
}

// ---------------------------------------------------------------------
// kernel 3: routed expert FFN, fused over H-chunks, bf16 WMMA
// grid = (B_TOK/TT, N_EXP), block = 512 (16 wave32)
// BF16W: B-fragments from pre-transposed bf16 weights (2x b128 / frag)
// ---------------------------------------------------------------------
template <bool BF16W>
__global__ void moe_expert_kernel(const float* __restrict__ x,
                                  const float* __restrict__ W1,
                                  const float* __restrict__ b1,
                                  const float* __restrict__ W2,
                                  const float* __restrict__ b2,
                                  const __bf16* __restrict__ W1T,
                                  const __bf16* __restrict__ W2T,
                                  const int* __restrict__ counts,
                                  const int* __restrict__ tok_list,
                                  const float* __restrict__ gate_list,
                                  float* __restrict__ y) {
    extern __shared__ char smem[];
    __bf16* ldsX   = (__bf16*)(smem + SM_X);
    __bf16* ldsH   = (__bf16*)(smem + SM_H);
    float*  red    = (float*)(smem + SM_RED);
    float*  rowred = (float*)(smem + SM_ROW);
    int*    tokS   = (int*)(smem + SM_TOK);
    float*  gateS  = (float*)(smem + SM_GATE);

    const int e   = blockIdx.y;
    const int cnt = counts[e];
    const int t0  = blockIdx.x * TT;
    if (t0 >= cnt) return;                 // block-uniform

    const int tid  = threadIdx.x;
    const int lane = tid & 31;
    const int wave = tid >> 5;
    const int half = lane >> 4;
    const int c    = lane & 15;

    if (tid < TT) {
        int slot = t0 + tid;
        if (slot < cnt) {
            tokS[tid]  = tok_list[e * B_TOK + slot];
            gateS[tid] = gate_list[e * B_TOK + slot];
        } else {
            tokS[tid]  = -1;
            gateS[tid] = 0.0f;
        }
    }
    __syncthreads();

    // stage x tile (gathered rows) into LDS as bf16; float4 loads, b64 stores
    for (int i = tid; i < TT * (D_IN / 4); i += BLK) {
        int row  = i >> 8;              // / (D_IN/4)
        int col4 = (i & (D_IN / 4 - 1)) * 4;
        int tk = tokS[row];
        v4f v = {0.f, 0.f, 0.f, 0.f};
        if (tk >= 0)
            v = *reinterpret_cast<const v4f*>(x + (size_t)tk * D_IN + col4);
        v4bf pk;
#pragma unroll
        for (int j = 0; j < 4; ++j) pk[j] = (__bf16)v[j];
        *reinterpret_cast<v4bf*>(&ldsX[(size_t)row * XS + col4]) = pk;
    }
    __syncthreads();

    const float*  W1e  = W1  + (size_t)e * D_IN * H_HID;
    const float*  W2e  = W2  + (size_t)e * H_HID * O_OUT;
    const __bf16* W1Te = W1T + (size_t)e * H_HID * D_IN;   // [H][D]
    const __bf16* W2Te = W2T + (size_t)e * O_OUT * H_HID;  // [O][H]
    const float*  b1e  = b1 + (size_t)e * H_HID;
    const float*  b2e  = b2 + (size_t)e * O_OUT;

    const v8f vzero = {0.f, 0.f, 0.f, 0.f, 0.f, 0.f, 0.f, 0.f};
    v8f acc[2][4];                     // 32 x 64 output slice per wave
#pragma unroll
    for (int m = 0; m < 2; ++m)
#pragma unroll
        for (int nf = 0; nf < 4; ++nf) acc[m][nf] = vzero;

    for (int hc = 0; hc < H_HID / HC; ++hc) {
        // hint-prefetch the next chunk's weights into GL2
        if (hc + 1 < H_HID / HC) {
            if (BF16W) {
                __builtin_prefetch(W1Te + (size_t)((hc + 1) * HC + wave * 16 + c) * D_IN, 0, 1);
                __builtin_prefetch(W2Te + (size_t)(wave * NCOLW + c) * H_HID + (hc + 1) * HC, 0, 1);
            } else {
                __builtin_prefetch(W1e + (size_t)(hc + 1) * HC + wave * 16 + c, 0, 1);
                __builtin_prefetch(W2e + (size_t)(hc + 1) * HC * O_OUT + wave * NCOLW + c, 0, 1);
            }
        }

        // ---- stage 1: h_chunk = relu(X(32x1024) @ W1[:, chunk](1024x256) + b1)
        // each of 16 waves owns 16 h columns: hcol = hc*HC + wave*16
        const int hcol = hc * HC + wave * 16;
        v8f h0 = vzero, h1 = vzero;
        for (int k0 = 0; k0 < D_IN; k0 += 32) {
            v16bf A0 = load_a_frag(ldsX, XS, 0,  k0, half, c);
            v16bf A1 = load_a_frag(ldsX, XS, 16, k0, half, c);
            v16bf Bf = BF16W
                ? load_b_frag_T(W1Te, D_IN, k0 + 16 * half, hcol + c)
                : load_b_frag(W1e, H_HID, k0 + 16 * half, hcol + c);
            h0 = wmma_bf16(A0, Bf, h0);
            h1 = wmma_bf16(A1, Bf, h1);
        }
        // bias + relu + bf16 -> LDS h tile
        {
            float bv = b1e[hcol + c];
#pragma unroll
            for (int r = 0; r < 8; ++r) {
                int row0 = half * 8 + r;
                int row1 = 16 + half * 8 + r;
                float f0 = h0[r] + bv; f0 = f0 > 0.0f ? f0 : 0.0f;
                float f1 = h1[r] + bv; f1 = f1 > 0.0f ? f1 : 0.0f;
                ldsH[(size_t)row0 * HS + wave * 16 + c] = (__bf16)f0;
                ldsH[(size_t)row1 * HS + wave * 16 + c] = (__bf16)f1;
            }
        }
        __syncthreads();

        // ---- stage 2: out(32x1024) += h_chunk(32x256) @ W2[chunk, :](256x1024)
        // each wave owns output columns [wave*64, wave*64+64)
        for (int k0 = 0; k0 < HC; k0 += 32) {
            v16bf A0 = load_a_frag(ldsH, HS, 0,  k0, half, c);
            v16bf A1 = load_a_frag(ldsH, HS, 16, k0, half, c);
            int krow = hc * HC + k0 + 16 * half;
#pragma unroll
            for (int nf = 0; nf < 4; ++nf) {
                int col = wave * NCOLW + nf * 16 + c;
                v16bf Bf = BF16W
                    ? load_b_frag_T(W2Te, H_HID, krow, col)
                    : load_b_frag(W2e, O_OUT, krow, col);
                acc[0][nf] = wmma_bf16(A0, Bf, acc[0][nf]);
                acc[1][nf] = wmma_bf16(A1, Bf, acc[1][nf]);
            }
        }
        __syncthreads();   // protect ldsH before next chunk overwrites it
    }

    // ---- add b2
    float b2v[4];
#pragma unroll
    for (int nf = 0; nf < 4; ++nf) b2v[nf] = b2e[wave * NCOLW + nf * 16 + c];
#pragma unroll
    for (int m = 0; m < 2; ++m)
#pragma unroll
        for (int nf = 0; nf < 4; ++nf)
#pragma unroll
            for (int r = 0; r < 8; ++r) acc[m][nf][r] += b2v[nf];

    // ---- row softmax over O=1024 (distributed across 16 waves) ----
    float pm[2][8];
#pragma unroll
    for (int m = 0; m < 2; ++m)
#pragma unroll
        for (int r = 0; r < 8; ++r) {
            float v = acc[m][0][r];
#pragma unroll
            for (int nf = 1; nf < 4; ++nf) v = fmaxf(v, acc[m][nf][r]);
            pm[m][r] = v;
        }
#pragma unroll
    for (int off = 8; off >= 1; off >>= 1)
#pragma unroll
        for (int m = 0; m < 2; ++m)
#pragma unroll
            for (int r = 0; r < 8; ++r)
                pm[m][r] = fmaxf(pm[m][r], __shfl_xor(pm[m][r], off, 32));
    if (c == 0) {
#pragma unroll
        for (int m = 0; m < 2; ++m)
#pragma unroll
            for (int r = 0; r < 8; ++r)
                red[(m * 16 + half * 8 + r) * NWAVE + wave] = pm[m][r];
    }
    __syncthreads();
    if (tid < TT) {
        float rm = red[tid * NWAVE];
#pragma unroll
        for (int w = 1; w < NWAVE; ++w) rm = fmaxf(rm, red[tid * NWAVE + w]);
        rowred[tid] = rm;
    }
    __syncthreads();

    // exp + partial sums
    float ps[2][8];
#pragma unroll
    for (int m = 0; m < 2; ++m)
#pragma unroll
        for (int r = 0; r < 8; ++r) {
            int row = m * 16 + half * 8 + r;
            float rm = rowred[row];
            float s = 0.0f;
#pragma unroll
            for (int nf = 0; nf < 4; ++nf) {
                float ev = __expf(acc[m][nf][r] - rm);
                acc[m][nf][r] = ev;
                s += ev;
            }
            ps[m][r] = s;
        }
#pragma unroll
    for (int off = 8; off >= 1; off >>= 1)
#pragma unroll
        for (int m = 0; m < 2; ++m)
#pragma unroll
            for (int r = 0; r < 8; ++r)
                ps[m][r] += __shfl_xor(ps[m][r], off, 32);
    __syncthreads();   // all reads of rowred(max) done before reuse as sum
    if (c == 0) {
#pragma unroll
        for (int m = 0; m < 2; ++m)
#pragma unroll
            for (int r = 0; r < 8; ++r)
                red[(m * 16 + half * 8 + r) * NWAVE + wave] = ps[m][r];
    }
    __syncthreads();
    if (tid < TT) {
        float rs = 0.0f;
#pragma unroll
        for (int w = 0; w < NWAVE; ++w) rs += red[tid * NWAVE + w];
        rowred[tid] = rs;
    }
    __syncthreads();

    // ---- gate-scaled scatter-add into y ----
#pragma unroll
    for (int m = 0; m < 2; ++m)
#pragma unroll
        for (int r = 0; r < 8; ++r) {
            int row = m * 16 + half * 8 + r;
            int tk = tokS[row];
            if (tk < 0) continue;
            float scale = gateS[row] / rowred[row];
            float* yr = y + (size_t)tk * O_OUT + wave * NCOLW + c;
#pragma unroll
            for (int nf = 0; nf < 4; ++nf)
                atomicAdd(yr + nf * 16, acc[m][nf][r] * scale);
        }
}

// ---------------------------------------------------------------------
extern "C" void kernel_launch(void* const* d_in, const int* in_sizes, int n_in,
                              void* d_out, int out_size, void* d_ws, size_t ws_size,
                              hipStream_t stream) {
    (void)in_sizes; (void)n_in; (void)out_size;
    const float* x  = (const float*)d_in[0];
    const float* wg = (const float*)d_in[1];
    const float* W1 = (const float*)d_in[2];
    const float* b1 = (const float*)d_in[3];
    const float* W2 = (const float*)d_in[4];
    const float* b2 = (const float*)d_in[5];
    // d_in[6] is k (== 2 per reference constants)

    float* y = (float*)d_out;   // y: [B_TOK*O_OUT], loss at index B_TOK*O_OUT

    char* ws = (char*)d_ws;
    int*    counts    = (int*)(ws + WS_COUNTS);
    float*  imp       = (float*)(ws + WS_IMP);
    float*  loadv     = (float*)(ws + WS_LOAD);
    int*    tok_list  = (int*)(ws + WS_TOK);
    float*  gate_list = (float*)(ws + WS_GATE);
    __bf16* w1t       = (__bf16*)(ws + WS_W1T);
    __bf16* w2t       = (__bf16*)(ws + WS_W2T);

    const bool use_bf16w = (ws_size >= (size_t)WS_NEEDED);

    moe_init_kernel<<<2048, 256, 0, stream>>>(y, B_TOK * O_OUT, counts, imp, loadv);
    moe_gate_kernel<<<B_TOK / 8, 256, 0, stream>>>(x, wg, counts, imp, loadv,
                                                   tok_list, gate_list);
    moe_loss_kernel<<<1, 32, 0, stream>>>(imp, loadv, y + (size_t)B_TOK * O_OUT);

    dim3 grid(B_TOK / TT, N_EXP);
    if (use_bf16w) {
        // W1 [E][D][H] f32 -> W1T [E][H][D] bf16 ; W2 [E][H][O] f32 -> W2T [E][O][H] bf16
        transpose_cvt_kernel<<<dim3(H_HID / 32, D_IN / 32, N_EXP), dim3(32, 8), 0, stream>>>(
            W1, w1t, D_IN, H_HID);
        transpose_cvt_kernel<<<dim3(O_OUT / 32, H_HID / 32, N_EXP), dim3(32, 8), 0, stream>>>(
            W2, w2t, H_HID, O_OUT);
        moe_expert_kernel<true><<<grid, BLK, SM_SIZE, stream>>>(
            x, W1, b1, W2, b2, w1t, w2t, counts, tok_list, gate_list, y);
    } else {
        moe_expert_kernel<false><<<grid, BLK, SM_SIZE, stream>>>(
            x, W1, b1, W2, b2, w1t, w2t, counts, tok_list, gate_list, y);
    }
}
